// BoundaryEnhance_68521908241133
// MI455X (gfx1250) — compile-verified, matrix operand order checked by reading, not run on hardware
//
#include <hip/hip_runtime.h>
#include <hip/hip_bf16.h>

// ---------------------------------------------------------------------------
// BoundaryEnhance fused kernel for MI455X (gfx1250, wave32, WMMA).
//
// Pipeline per pixel: Sobel-x/y depthwise 3x3 -> 768-vec "edges",
// fused = W[384x768] @ edges + b, LayerNorm over channels, exact GELU,
// residual add.  The 87 GFLOP fusion GEMM is mapped onto
// v_wmma_f32_16x16x32_bf16 (bf16 A/B, f32 accum) which puts the kernel at the
// roofline knee (~35us compute vs ~19us HBM floor); fp32 WMMA (16x16x4) would
// be 8x slower, fp8 gains nothing past the memory floor.
// ---------------------------------------------------------------------------

typedef __attribute__((ext_vector_type(16))) __bf16 v16bf;
typedef __attribute__((ext_vector_type(8)))  float  v8f;

#define BATCH   16
#define CH      384          // output channels
#define K2      768          // 2*CH contraction depth
#define IMH     96
#define IMW     96
#define MT      16           // pixels per workgroup (one row segment)
#define KTILES  24           // 768 / 32
#define NTILES  24           // 384 / 16

struct __attribute__((aligned(16))) U4 { unsigned int x, y, z, w; };
struct __attribute__((aligned(32))) Frag32 { U4 lo, hi; };

__device__ __forceinline__ v16bf frag_cast(Frag32 f) {
    return __builtin_bit_cast(v16bf, f);
}

// round-to-nearest-even f32 -> bf16, packed pair into one dword
__device__ __forceinline__ unsigned int pack_bf16x2(float a, float b) {
    unsigned int ua = __builtin_bit_cast(unsigned int, a);
    unsigned int ub = __builtin_bit_cast(unsigned int, b);
    ua = (ua + 0x7FFFu + ((ua >> 16) & 1u)) >> 16;
    ub = (ub + 0x7FFFu + ((ub >> 16) & 1u)) >> 16;
    return (ua & 0xFFFFu) | (ub << 16);
}

// ---------------------------------------------------------------------------
// Prep: fusion_w (f32 [n=384][k=768]) -> bf16 B-fragments, one dword per
// (ktile, ntile, lane, j) so each lane's 32x16 B tile is 32 contiguous bytes
// (two global_load_b128).  B[k][n] = fusion_w[n][k]; lane selects n (lane&15),
// (j, lane>>4) select the packed K pair per ISA 16-bit fragment striping.
// ---------------------------------------------------------------------------
__global__ __launch_bounds__(256) void wpack_kernel(
    const float* __restrict__ fw, unsigned int* __restrict__ wp) {
    int idx  = blockIdx.x * 256 + threadIdx.x;     // < 24*24*32*8 = 147456
    int kt   = idx / 6144;
    int rem  = idx - kt * 6144;
    int nt   = rem >> 8;
    int r2   = rem & 255;
    int lane = r2 >> 3;
    int j    = r2 & 7;
    int n    = nt * 16 + (lane & 15);
    int k    = kt * 32 + (j >> 2) * 16 + ((lane >> 4) << 3) + (j & 3) * 2;
    const float* row = fw + n * K2 + k;
    wp[idx] = pack_bf16x2(row[0], row[1]);
}

// ---------------------------------------------------------------------------
// Main fused kernel: one workgroup = 16 consecutive pixels of one image row.
// ---------------------------------------------------------------------------
__global__ __launch_bounds__(256) void boundary_enhance_kernel(
    const float* __restrict__ x, const unsigned int* __restrict__ wpack,
    const float* __restrict__ fb, const float* __restrict__ lnw,
    const float* __restrict__ lnb, float* __restrict__ out) {
    extern __shared__ float smem[];
    float*        xtile  = smem;                                  // 3*18*384 f32
    unsigned int* edgesA = (unsigned int*)(smem + 3 * 18 * CH);   // 6144 dwords
    float*        fused  = smem + 3 * 18 * CH + 6144;             // 16*384 f32
    float*        psum   = (float*)edgesA;                        // reused post-GEMM
    float*        psq    = psum + 256;

    const int tid = threadIdx.x;
    const int bx  = blockIdx.x;
    const int wt  = bx % (IMW / MT);
    const int h   = (bx / (IMW / MT)) % IMH;
    const int b   = bx / ((IMW / MT) * IMH);
    const int w0  = wt * MT;

    // ---- stage x halo tile: 3 rows x 18 cols x 384 ch, SAME zero padding ----
    for (int idx = tid; idx < 3 * 18 * CH; idx += 256) {
        int c   = idx / 54;
        int rem = idx - c * 54;
        int r   = rem / 18;
        int col = rem - r * 18;
        int hh  = h + r - 1;
        int ww  = w0 + col - 1;
        float v = 0.0f;
        if (hh >= 0 && hh < IMH && ww >= 0 && ww < IMW)
            v = x[(((long)b * CH + c) * IMH + hh) * IMW + ww];
        xtile[idx] = v;
    }
    __syncthreads();

    // ---- Sobel edges -> bf16, written directly in A-fragment dword layout ----
    // slot (kt, lane, j): m = lane&15, k = kt*32 + (j>>2)*16 + (lane>>4)*8 + (j&3)*2
    for (int i = 0; i < 24; ++i) {
        int idx  = tid + i * 256;            // < 6144
        int kt   = idx >> 8;
        int rem  = idx & 255;
        int lane = rem >> 3;
        int j    = rem & 7;
        int m    = lane & 15;
        int k0   = kt * 32 + (j >> 2) * 16 + ((lane >> 4) << 3) + (j & 3) * 2;
        float e[2];
#pragma unroll
        for (int p = 0; p < 2; ++p) {
            int k = k0 + p;
            int c = (k < CH) ? k : (k - CH);
            const float* t = xtile + c * 54;
            int col = m + 1;
            float a00 = t[col - 1],      a01 = t[col],      a02 = t[col + 1];
            float a10 = t[18 + col - 1],                    a12 = t[18 + col + 1];
            float a20 = t[36 + col - 1], a21 = t[36 + col], a22 = t[36 + col + 1];
            float ex = (a02 - a00) + 2.0f * (a12 - a10) + (a22 - a20);
            float ey = (a20 + 2.0f * a21 + a22) - (a00 + 2.0f * a01 + a02);
            e[p] = (k < CH) ? ex : ey;
        }
        edgesA[idx] = pack_bf16x2(e[0], e[1]);
    }
    __syncthreads();

    // ---- GEMM: M=16 x N=384 x K=768 via v_wmma_f32_16x16x32_bf16 ----------
    const int wave = tid >> 5;
    const int lane = tid & 31;
    v8f acc0 = {}, acc1 = {}, acc2 = {};
    {
        const Frag32* bgl = (const Frag32*)wpack;
        const int nt0 = wave * 3;
#pragma unroll
        for (int kt = 0; kt < KTILES; ++kt) {
            v16bf a = frag_cast(*(const Frag32*)(edgesA + (kt * 32 + lane) * 8));
            const Frag32* bp = bgl + (kt * NTILES + nt0) * 32 + lane;
            v16bf b0 = frag_cast(bp[0]);
            v16bf b1 = frag_cast(bp[32]);
            v16bf b2 = frag_cast(bp[64]);
            acc0 = __builtin_amdgcn_wmma_f32_16x16x32_bf16(
                false, a, false, b0, (short)0, acc0, false, false);
            acc1 = __builtin_amdgcn_wmma_f32_16x16x32_bf16(
                false, a, false, b1, (short)0, acc1, false, false);
            acc2 = __builtin_amdgcn_wmma_f32_16x16x32_bf16(
                false, a, false, b2, (short)0, acc2, false, false);
        }
    }

    // ---- spill accumulators (+fusion bias) to LDS fused[m][n] -------------
    {
        const int mhi = (lane >> 4) * 8;     // C/D layout: vgpr r -> M = r (+8 hi lanes)
        const int nl  = lane & 15;
#pragma unroll
        for (int t = 0; t < 3; ++t) {
            int n = (wave * 3 + t) * 16 + nl;
            float bias = fb[n];
            v8f acc = (t == 0) ? acc0 : ((t == 1) ? acc1 : acc2);
#pragma unroll
            for (int r = 0; r < 8; ++r)
                fused[(mhi + r) * CH + n] = acc[r] + bias;
        }
    }
    __syncthreads();

    // ---- LayerNorm over channels + exact GELU + residual ------------------
    {
        const int p = tid & 15;     // pixel (consecutive w -> coalesced stores)
        const int s = tid >> 4;     // channel slice, 24 channels each
        const int cbase = s * 24;
        float sum = 0.0f, sq = 0.0f;
#pragma unroll 4
        for (int i = 0; i < 24; ++i) {
            float v = fused[p * CH + cbase + i];
            sum += v; sq += v * v;
        }
        psum[p * 16 + s] = sum;
        psq[p * 16 + s]  = sq;
        __syncthreads();
        float ts = 0.0f, tq = 0.0f;
#pragma unroll
        for (int s2 = 0; s2 < 16; ++s2) {
            ts += psum[p * 16 + s2];
            tq += psq[p * 16 + s2];
        }
        float mu  = ts * (1.0f / CH);
        float var = tq * (1.0f / CH) - mu * mu;
        float rs  = rsqrtf(var + 1e-5f);

        float* op = out + (((long)b * CH) * IMH + h) * IMW + (w0 + p);
        const float* xc = xtile + 18 + (p + 1);     // center row/col, stride 54/ch
        for (int i = 0; i < 24; ++i) {
            int c = cbase + i;
            float v  = fused[p * CH + c];
            float nr = (v - mu) * rs * lnw[c] + lnb[c];
            float g  = 0.5f * nr * (1.0f + erff(nr * 0.70710678118654752f));
            op[(long)c * (IMH * IMW)] = xc[c * 54] + g;
        }
    }
}

extern "C" void kernel_launch(void* const* d_in, const int* in_sizes, int n_in,
                              void* d_out, int out_size, void* d_ws, size_t ws_size,
                              hipStream_t stream) {
    const float* x   = (const float*)d_in[0];
    const float* fw  = (const float*)d_in[1];   // (384, 768)
    const float* fb  = (const float*)d_in[2];
    const float* lnw = (const float*)d_in[3];
    const float* lnb = (const float*)d_in[4];
    unsigned int* wpack = (unsigned int*)d_ws;  // 147456 dwords = 576 KB, fully written below

    // pack fusion weights to bf16 WMMA B-fragments (stream-ordered before main)
    wpack_kernel<<<576, 256, 0, stream>>>(fw, wpack);

    const size_t smem = (size_t)(3 * 18 * CH + 6144 + 16 * CH) * sizeof(float); // 132096 B
    (void)hipFuncSetAttribute((const void*)boundary_enhance_kernel,
                              hipFuncAttributeMaxDynamicSharedMemorySize, (int)smem);
    const int nblocks = BATCH * IMH * (IMW / MT);   // 9216
    boundary_enhance_kernel<<<nblocks, 256, smem, stream>>>(
        x, wpack, fb, lnw, lnb, (float*)d_out);
}
